// BahdanauAttentionAudio_19945828122625
// MI455X (gfx1250) — compile-verified, matrix operand order checked by reading, not run on hardware
//
#include <hip/hip_runtime.h>
#include <hip/hip_bf16.h>
#include <cstdint>

// ---------------- problem constants ----------------
#define B_   32
#define T_   2048
#define H_   1024
#define U_   1024
#define KN_  32
#define KS_  15

typedef __attribute__((ext_vector_type(16))) __bf16 v16bf;
typedef __attribute__((ext_vector_type(8)))  float  v8f;

// round-to-nearest-even f32 -> bf16 (bit pattern), avoids scalar __bf16 math
static __device__ __forceinline__ unsigned short f2bf(float x) {
    unsigned u = __builtin_bit_cast(unsigned, x);
    unsigned rounding = 0x7FFFu + ((u >> 16) & 1u);
    u += rounding;
    return (unsigned short)(u >> 16);
}

// branch-free tanh: one v_exp_f32 + one v_rcp_f32, clamped so e never overflows
static __device__ __forceinline__ float fast_tanh(float x) {
    float xc = fminf(fmaxf(x, -10.f), 10.f);
    float e  = __expf(2.0f * xc);
    return (e - 1.0f) * __builtin_amdgcn_rcpf(e + 1.0f);
}

// ---------------- prep: f32 weights -> bf16 workspace ----------------
__global__ __launch_bounds__(256)
void prep_kernel(const float* __restrict__ W1_w, const float* __restrict__ lpw,
                 unsigned short* __restrict__ W1bf, unsigned short* __restrict__ lpbf) {
    size_t i = (size_t)blockIdx.x * 256 + threadIdx.x;
    size_t stride = (size_t)gridDim.x * 256;
    size_t n1 = (size_t)U_ * H_;
    for (size_t j = i; j < n1; j += stride) W1bf[j] = f2bf(W1_w[j]);
    size_t n2 = (size_t)U_ * KN_;
    for (size_t j = i; j < n2; j += stride) lpbf[j] = f2bf(lpw[j]);
}

// ---------------- bias(b,u) = q(b,:)·W2_w(u,:) + W1_b + W2_b ----------------
__global__ __launch_bounds__(256)
void bias_kernel(const float* __restrict__ q, const float* __restrict__ W2_w,
                 const float* __restrict__ W1_b, const float* __restrict__ W2_b,
                 float* __restrict__ bias) {
    int idx = blockIdx.x * 256 + threadIdx.x;      // B_*U_ = 32768
    int b = idx >> 10, u = idx & (U_ - 1);
    const float* qr = q + (size_t)b * H_;
    const float* wr = W2_w + (size_t)u * H_;
    float acc = 0.f;
    #pragma unroll 4
    for (int h = 0; h < H_; ++h) acc += qr[h] * wr[h];
    bias[idx] = acc + W1_b[u] + W2_b[u];
}

// ---------------- conv1d, output transposed (b,t,k) as bf16 ----------------
__global__ __launch_bounds__(256)
void conv_kernel(const float* __restrict__ prev_att, const float* __restrict__ conv_w,
                 unsigned short* __restrict__ locT) {
    int idx = blockIdx.x * 256 + threadIdx.x;      // B_*T_*KN_ = 2097152
    int k = idx & (KN_ - 1);
    int t = (idx >> 5) & (T_ - 1);
    int b = idx >> 16;
    const float* pa = prev_att + (size_t)b * T_;
    const float* cw = conv_w + (size_t)k * (2 * KS_ + 1);
    float acc = 0.f;
    #pragma unroll
    for (int j = 0; j < 2 * KS_ + 1; ++j) {
        int tt = t + j - KS_;
        float v = (tt >= 0 && tt < T_) ? pa[tt] : 0.f;
        acc += v * cw[j];
    }
    locT[idx] = f2bf(acc);
}

// ---------------- fused main kernel ----------------
// Block: 256 threads (8 waves), owns M=64 (b,t) rows staged as bf16 in LDS.
// Wave w handles u-tile groups g = w, w+8; each group is N=64 (4 u-tiles).
// Register tile per wave: 4 (M-subtiles) x 4 (u-tiles) C accumulators, so each
// A fragment (LDS) and B fragment (L2-resident W1) feeds 4 WMMAs.
#define SROWB 2064                 // 1024 bf16 + 16 B pad per LDS row
#define MROWS 64
#define LDS_BYTES (MROWS * SROWB + MROWS * 4)

__global__ __launch_bounds__(256)
void attn_score_kernel(const float* __restrict__ values,
                       const unsigned short* __restrict__ W1bf,
                       const unsigned short* __restrict__ lpbf,
                       const unsigned short* __restrict__ locT,
                       const float* __restrict__ bias,
                       const float* __restrict__ V_w,
                       const float* __restrict__ V_b,
                       float* __restrict__ score_out) {
    extern __shared__ __align__(16) unsigned char smem[];
    unsigned char* ldsV = smem;                          // MROWS x SROWB
    float* sacc = (float*)(smem + MROWS * SROWB);        // MROWS partial scores

    const int tid = threadIdx.x;
    const int bid = blockIdx.x;                 // 1024 = B_*(T_/64)
    const int b   = bid >> 5;
    const int t0  = (bid & 31) << 6;
    const size_t row0 = (size_t)b * T_ + t0;

    // stage 64x1024 f32 values rows as bf16 into LDS
    for (int i = tid; i < MROWS * 512; i += 256) {  // 512 float-pairs per row
        int r = i >> 9, p = i & 511;
        float2 f = *(const float2*)(values + (row0 + r) * H_ + (size_t)p * 2);
        unsigned lo = f2bf(f.x), hi = f2bf(f.y);
        *(unsigned*)(ldsV + r * SROWB + p * 4) = lo | (hi << 16);
    }
    if (tid < MROWS) sacc[tid] = 0.f;
    __syncthreads();

    const int lane = tid & 31;
    const int wv   = tid >> 5;
    const int half = lane >> 4;     // K-half select per WMMA 16-bit layout
    const int n    = lane & 15;     // A row-in-subtile / B column index
    const int h8   = half * 8;

    union Frag { v16bf v; uint4 q[2]; };

    for (int g = wv; g < 16; g += 8) {          // two u-groups per wave
        const int u0 = g << 6;                  // group covers u0..u0+63
        v8f c[4][4];
        #pragma unroll
        for (int i = 0; i < 4; ++i)
            #pragma unroll
            for (int j = 0; j < 4; ++j)
                c[i][j] = (v8f){0.f,0.f,0.f,0.f,0.f,0.f,0.f,0.f};

        const unsigned char* aB = ldsV + n * SROWB + h8 * 2;
        const unsigned short* bB = W1bf + (size_t)(u0 + n) * H_ + h8;

        for (int k0 = 0; k0 < H_; k0 += 32) {
            v16bf af[4], bv[4];
            #pragma unroll
            for (int i = 0; i < 4; ++i) {
                Frag a;
                const unsigned char* p = aB + i * (16 * SROWB) + (size_t)k0 * 2;
                a.q[0] = *(const uint4*)(p);
                a.q[1] = *(const uint4*)(p + 32);
                af[i] = a.v;
            }
            #pragma unroll
            for (int j = 0; j < 4; ++j) {
                Frag bm;
                const unsigned short* p = bB + (size_t)j * (16 * H_) + k0;
                bm.q[0] = *(const uint4*)(p);
                bm.q[1] = *(const uint4*)(p + 16);
                bv[j] = bm.v;
            }
            #pragma unroll
            for (int i = 0; i < 4; ++i)
                #pragma unroll
                for (int j = 0; j < 4; ++j)
                    c[i][j] = __builtin_amdgcn_wmma_f32_16x16x32_bf16(
                        false, af[i], false, bv[j], (short)0, c[i][j], false, false);
        }

        // location-context: one extra K=32 WMMA per C tile
        {
            v16bf af[4], bv[4];
            #pragma unroll
            for (int i = 0; i < 4; ++i) {
                Frag a;
                const unsigned short* p = locT + (row0 + 16 * i + n) * KN_ + h8;
                a.q[0] = *(const uint4*)(p);
                a.q[1] = *(const uint4*)(p + 16);
                af[i] = a.v;
            }
            #pragma unroll
            for (int j = 0; j < 4; ++j) {
                Frag bm;
                const unsigned short* p = lpbf + (size_t)(u0 + 16 * j + n) * KN_ + h8;
                bm.q[0] = *(const uint4*)(p);
                bm.q[1] = *(const uint4*)(p + 16);
                bv[j] = bm.v;
            }
            #pragma unroll
            for (int i = 0; i < 4; ++i)
                #pragma unroll
                for (int j = 0; j < 4; ++j)
                    c[i][j] = __builtin_amdgcn_wmma_f32_16x16x32_bf16(
                        false, af[i], false, bv[j], (short)0, c[i][j], false, false);
        }

        // epilogue: + bias, tanh, dot with V_w, reduce columns, accumulate rows
        float bias_v[4], vw[4];
        #pragma unroll
        for (int j = 0; j < 4; ++j) {
            bias_v[j] = bias[b * U_ + u0 + 16 * j + n];
            vw[j]     = V_w[u0 + 16 * j + n];
        }
        #pragma unroll
        for (int i = 0; i < 4; ++i) {
            float part[8];
            #pragma unroll
            for (int r = 0; r < 8; ++r) part[r] = 0.f;
            #pragma unroll
            for (int j = 0; j < 4; ++j)
                #pragma unroll
                for (int r = 0; r < 8; ++r)
                    part[r] += fast_tanh(c[i][j][r] + bias_v[j]) * vw[j];
            #pragma unroll
            for (int r = 0; r < 8; ++r) {        // reduce 16 columns per half
                float v = part[r];
                v += __shfl_xor(v, 1, 32);
                v += __shfl_xor(v, 2, 32);
                v += __shfl_xor(v, 4, 32);
                v += __shfl_xor(v, 8, 32);
                if (n == 0) atomicAdd(&sacc[16 * i + h8 + r], v);
            }
        }
    }
    __syncthreads();
    if (tid < MROWS) score_out[row0 + tid] = sacc[tid] + V_b[0];
}

// ---------------- softmax over T per batch ----------------
__global__ __launch_bounds__(256)
void softmax_kernel(const float* __restrict__ score, float* __restrict__ att) {
    __shared__ float red[256];
    const int b = blockIdx.x, tid = threadIdx.x;
    const float* s = score + (size_t)b * T_;
    float loc[8], mx = -1e30f;
    #pragma unroll
    for (int i = 0; i < 8; ++i) { loc[i] = s[tid + i * 256]; mx = fmaxf(mx, loc[i]); }
    red[tid] = mx; __syncthreads();
    for (int st = 128; st > 0; st >>= 1) {
        if (tid < st) red[tid] = fmaxf(red[tid], red[tid + st]);
        __syncthreads();
    }
    mx = red[0]; __syncthreads();
    float sum = 0.f;
    #pragma unroll
    for (int i = 0; i < 8; ++i) { loc[i] = __expf(loc[i] - mx); sum += loc[i]; }
    red[tid] = sum; __syncthreads();
    for (int st = 128; st > 0; st >>= 1) {
        if (tid < st) red[tid] += red[tid + st];
        __syncthreads();
    }
    const float inv = 1.f / red[0];
    float* a = att + (size_t)b * T_;
    #pragma unroll
    for (int i = 0; i < 8; ++i) a[tid + i * 256] = loc[i] * inv;
}

// ---------------- context(b,h) = sum_t att(b,t)*values(b,t,h) ----------------
__global__ __launch_bounds__(256)
void context_kernel(const float* __restrict__ values, const float* __restrict__ att,
                    float* __restrict__ ctx) {
    __shared__ float la[256];
    const int b = blockIdx.x >> 2;
    const int h = ((blockIdx.x & 3) << 8) + threadIdx.x;
    float acc = 0.f;
    for (int tb = 0; tb < T_; tb += 256) {
        __syncthreads();
        la[threadIdx.x] = att[(size_t)b * T_ + tb + threadIdx.x];
        __syncthreads();
        const float* vp = values + ((size_t)b * T_ + tb) * H_ + h;
        #pragma unroll 8
        for (int tt = 0; tt < 256; ++tt) acc += la[tt] * vp[(size_t)tt * H_];
    }
    ctx[(size_t)b * H_ + h] = acc;
}

// ---------------- launcher ----------------
extern "C" void kernel_launch(void* const* d_in, const int* in_sizes, int n_in,
                              void* d_out, int out_size, void* d_ws, size_t ws_size,
                              hipStream_t stream) {
    const float* query  = (const float*)d_in[0];
    const float* values = (const float*)d_in[1];
    const float* prev   = (const float*)d_in[2];
    const float* W1_w   = (const float*)d_in[3];
    const float* W1_b   = (const float*)d_in[4];
    const float* W2_w   = (const float*)d_in[5];
    const float* W2_b   = (const float*)d_in[6];
    const float* V_w    = (const float*)d_in[7];
    const float* V_b    = (const float*)d_in[8];
    const float* conv_w = (const float*)d_in[9];
    const float* lpw    = (const float*)d_in[10];

    float* out   = (float*)d_out;
    float* ctx   = out;                       // (B,H)   32768
    float* att   = out + 32768;               // (B,T,1) 65536
    float* score = out + 32768 + 65536;       // (B,T,1) 65536

    unsigned char* ws = (unsigned char*)d_ws;
    unsigned short* W1bf = (unsigned short*)(ws);                       // 2 MB
    unsigned short* lpbf = (unsigned short*)(ws + 2097152);             // 64 KB
    unsigned short* locT = (unsigned short*)(ws + 2097152 + 65536);     // 4 MB
    float*          bias = (float*)(ws + 2097152 + 65536 + 4194304);    // 128 KB

    prep_kernel<<<dim3(4096), dim3(256), 0, stream>>>(W1_w, lpw, W1bf, lpbf);
    bias_kernel<<<dim3(128),  dim3(256), 0, stream>>>(query, W2_w, W1_b, W2_b, bias);
    conv_kernel<<<dim3(8192), dim3(256), 0, stream>>>(prev, conv_w, locT);
    attn_score_kernel<<<dim3(1024), dim3(256), LDS_BYTES, stream>>>(
        values, W1bf, lpbf, locT, bias, V_w, V_b, score);
    softmax_kernel<<<dim3(32), dim3(256), 0, stream>>>(score, att);
    context_kernel<<<dim3(128), dim3(256), 0, stream>>>(values, att, ctx);
}
